// GATLayer_154618823051
// MI455X (gfx1250) — matrix operand order repaired and checked by reading, compiled, not run on hardware
//
#include <hip/hip_runtime.h>
#include <math.h>

typedef __attribute__((ext_vector_type(16))) _Float16 v16h;
typedef __attribute__((ext_vector_type(8)))  _Float16 v8h;
typedef __attribute__((ext_vector_type(8)))  float    v8f;

#define GAT_N    50000
#define GAT_E    1600000
#define GAT_F    128
#define GAT_ALPHA 0.2f
#define LDS_PITCH 136   // halves per row in LDS (128 + 8 pad) -> 272B, conflict-free

// ---------------------------------------------------------------------------
// Kernel 1: Wt[n*128 + k] = (f16) W[k*128 + n]   (column-major f16 copy of W)
// ---------------------------------------------------------------------------
__global__ void gat_wt_kernel(const float* __restrict__ W, _Float16* __restrict__ Wt) {
    int idx = blockIdx.x * blockDim.x + threadIdx.x;   // 0 .. 16383
    int n = idx >> 7;          // column of W
    int k = idx & 127;         // row of W
    Wt[n * GAT_F + k] = (_Float16)W[k * GAT_F + n];
}

// ---------------------------------------------------------------------------
// Kernel 2: hw = h @ W via v_wmma_f32_16x16x32_f16.
// Wt (f16, column-major) is staged into LDS once per block via the CDNA5
// async load-to-LDS path; WMMA B fragments are then fed from ds_load_b128.
// Block = 256 threads = 8 waves; wave w computes rows [blk*128 + w*16, +16).
// ---------------------------------------------------------------------------
__global__ void gat_gemm_kernel(const float* __restrict__ h,
                                const _Float16* __restrict__ Wt,
                                float* __restrict__ hw) {
    __shared__ _Float16 Bs[GAT_F * LDS_PITCH];   // 128 rows x 136 halves = 34 KB

    // ---- async stage Wt -> LDS (2048 x 16B chunks, 8 per thread) ----
    {
        const int t = threadIdx.x;
#pragma unroll
        for (int i = 0; i < 8; ++i) {
            int g = i * 256 + t;            // 0..2047
            int r = g >> 4;                 // Wt row (= W column) 0..127
            int c = g & 15;                 // 16B chunk within row
            unsigned lds_off = (unsigned)(uintptr_t)&Bs[r * LDS_PITCH + c * 8];
            unsigned goff    = (unsigned)(r * 256 + c * 16);   // bytes into Wt
            asm volatile("global_load_async_to_lds_b128 %0, %1, %2 offset:0"
                         :: "v"(lds_off), "v"(goff), "s"(Wt)
                         : "memory");
        }
        asm volatile("s_wait_asynccnt 0x0" ::: "memory");
    }
    __syncthreads();

    const int wid  = threadIdx.x >> 5;
    const int lane = threadIdx.x & 31;
    const int row0 = blockIdx.x * 128 + wid * 16;
    const int mlo  = lane & 15;
    const int hi   = (lane >> 4);           // 0 for lanes 0-15, 1 for 16-31

    // clamp A row for tail tiles (stores are guarded below)
    int rA = row0 + mlo;
    if (rA >= GAT_N) rA = GAT_N - 1;
    const float* Arow = h + (size_t)rA * GAT_F;

    v8f acc[8];
#pragma unroll
    for (int nt = 0; nt < 8; ++nt) acc[nt] = (v8f){0.f,0.f,0.f,0.f,0.f,0.f,0.f,0.f};

#pragma unroll
    for (int kc = 0; kc < 4; ++kc) {
        // ---- A fragment: 16-bit A 16x32 layout.
        // lane<16: K = kc*32 + {0..7, 16..23}; lane>=16: K = kc*32 + {8..15, 24..31}
        const int kb = kc * 32 + hi * 8;
        float av[16];
        *(float4*)&av[0]  = *(const float4*)(Arow + kb);
        *(float4*)&av[4]  = *(const float4*)(Arow + kb + 4);
        *(float4*)&av[8]  = *(const float4*)(Arow + kb + 16);
        *(float4*)&av[12] = *(const float4*)(Arow + kb + 20);
        v16h a;
#pragma unroll
        for (int j = 0; j < 16; ++j) a[j] = (_Float16)av[j];

        // ---- B fragments from LDS: 16-bit B 32x16 layout.
        // lane<16: K = kc*32 + {0..7, 16..23} for column N=lane; lane>=16: K = kc*32 + {8..15, 24..31}
        const int kb2 = kc * 32 + hi * 8;
#pragma unroll
        for (int nt = 0; nt < 8; ++nt) {
            const _Float16* Bp = &Bs[(nt * 16 + mlo) * LDS_PITCH + kb2];
            v8h b0 = *(const v8h*)Bp;        // ds_load_b128
            v8h b1 = *(const v8h*)(Bp + 16); // ds_load_b128
            v16h b;
#pragma unroll
            for (int j = 0; j < 8; ++j) { b[j] = b0[j]; b[8 + j] = b1[j]; }
            acc[nt] = __builtin_amdgcn_wmma_f32_16x16x32_f16(
                false, a, false, b, (short)0, acc[nt], false, false);
        }
    }

    // ---- Store C: 32-bit 16x16 layout: VGPR r -> M = r + 8*hi, N = lane&15
#pragma unroll
    for (int nt = 0; nt < 8; ++nt) {
#pragma unroll
        for (int r = 0; r < 8; ++r) {
            int m = row0 + hi * 8 + r;
            if (m < GAT_N) hw[(size_t)m * GAT_F + nt * 16 + mlo] = acc[nt][r];
        }
    }
}

// ---------------------------------------------------------------------------
// Kernel 3: s_src[i] = hw[i,:]·a_src ; s_dst[i] = hw[i,:]·a_dst
// one wave per node, 4 features per lane, shfl_xor wave reduction
// ---------------------------------------------------------------------------
__global__ void gat_scores_kernel(const float* __restrict__ hw,
                                  const float* __restrict__ a_src,
                                  const float* __restrict__ a_dst,
                                  float* __restrict__ s_src,
                                  float* __restrict__ s_dst) {
    int node = blockIdx.x * 8 + (threadIdx.x >> 5);
    int lane = threadIdx.x & 31;
    if (node >= GAT_N) return;
    float4 x  = ((const float4*)(hw + (size_t)node * GAT_F))[lane];
    float4 as = ((const float4*)a_src)[lane];
    float4 ad = ((const float4*)a_dst)[lane];
    float ps = x.x*as.x + x.y*as.y + x.z*as.z + x.w*as.w;
    float pd = x.x*ad.x + x.y*ad.y + x.z*ad.z + x.w*ad.w;
#pragma unroll
    for (int off = 16; off >= 1; off >>= 1) {
        ps += __shfl_xor(ps, off, 32);
        pd += __shfl_xor(pd, off, 32);
    }
    if (lane == 0) { s_src[node] = ps; s_dst[node] = pd; }
}

// ---------------------------------------------------------------------------
// Kernel 4: row_ptr[i] = lower_bound(row, i)  (row is sorted); row_ptr[N] = E
// ---------------------------------------------------------------------------
__global__ void gat_rowptr_kernel(const int* __restrict__ row, int* __restrict__ row_ptr) {
    int i = blockIdx.x * blockDim.x + threadIdx.x;
    if (i > GAT_N) return;
    int lo = 0, hiE = GAT_E;
    while (lo < hiE) {
        int mid = (lo + hiE) >> 1;
        if (row[mid] < i) lo = mid + 1; else hiE = mid;
    }
    row_ptr[i] = lo;
}

// ---------------------------------------------------------------------------
// Kernel 5: fused leaky-relu + online segment softmax + weighted gather + ELU
// one wave per destination node; each lane owns 4 features (float4)
// ---------------------------------------------------------------------------
__global__ void gat_aggregate_kernel(const float* __restrict__ hw,
                                     const float* __restrict__ s_src,
                                     const float* __restrict__ s_dst,
                                     const int* __restrict__ row_ptr,
                                     const int* __restrict__ col,
                                     float* __restrict__ out) {
    int node = blockIdx.x * 8 + (threadIdx.x >> 5);
    int lane = threadIdx.x & 31;
    if (node >= GAT_N) return;

    const int start = row_ptr[node];
    const int end   = row_ptr[node + 1];
    const float si  = s_src[node];

    float4 acc = {0.f, 0.f, 0.f, 0.f};
    float m = -INFINITY;
    float d = 0.f;

    for (int j = start; j < end; ++j) {
        int c = col[j];
        float e = si + s_dst[c];
        e = (e > 0.f) ? e : GAT_ALPHA * e;
        float mn    = fmaxf(m, e);
        float scale = __expf(m - mn);   // 0 on first edge (m = -inf)
        float w     = __expf(e - mn);
        float4 hv = ((const float4*)(hw + (size_t)c * GAT_F))[lane];
        d = d * scale + w;
        acc.x = acc.x * scale + w * hv.x;
        acc.y = acc.y * scale + w * hv.y;
        acc.z = acc.z * scale + w * hv.z;
        acc.w = acc.w * scale + w * hv.w;
        m = mn;
    }

    float inv = (d > 0.f) ? (1.f / d) : 0.f;
    float4 o;
    o.x = acc.x * inv; o.y = acc.y * inv; o.z = acc.z * inv; o.w = acc.w * inv;
    // ELU (alpha = 1)
    o.x = (o.x > 0.f) ? o.x : (__expf(o.x) - 1.f);
    o.y = (o.y > 0.f) ? o.y : (__expf(o.y) - 1.f);
    o.z = (o.z > 0.f) ? o.z : (__expf(o.z) - 1.f);
    o.w = (o.w > 0.f) ? o.w : (__expf(o.w) - 1.f);
    ((float4*)(out + (size_t)node * GAT_F))[lane] = o;
}

// ---------------------------------------------------------------------------
extern "C" void kernel_launch(void* const* d_in, const int* in_sizes, int n_in,
                              void* d_out, int out_size, void* d_ws, size_t ws_size,
                              hipStream_t stream) {
    const float* h     = (const float*)d_in[0];
    const float* W     = (const float*)d_in[1];
    const float* a_src = (const float*)d_in[2];
    const float* a_dst = (const float*)d_in[3];
    const int*   row   = (const int*)d_in[4];
    const int*   col   = (const int*)d_in[5];
    float* out = (float*)d_out;

    // workspace carve-up
    float* hw    = (float*)d_ws;                       // N*128 f32
    float* s_src = hw + (size_t)GAT_N * GAT_F;         // N f32
    float* s_dst = s_src + GAT_N;                      // N f32
    int*   row_ptr = (int*)(s_dst + GAT_N);            // N+1 i32
    uintptr_t p = (uintptr_t)(row_ptr + GAT_N + 1);
    p = (p + 63) & ~(uintptr_t)63;                     // 64B align
    _Float16* Wt = (_Float16*)p;                       // 128*128 f16

    gat_wt_kernel<<<dim3((GAT_F * GAT_F) / 256), dim3(256), 0, stream>>>(W, Wt);

    gat_gemm_kernel<<<dim3((GAT_N + 127) / 128), dim3(256), 0, stream>>>(h, Wt, hw);

    gat_scores_kernel<<<dim3((GAT_N + 7) / 8), dim3(256), 0, stream>>>(
        hw, a_src, a_dst, s_src, s_dst);

    gat_rowptr_kernel<<<dim3((GAT_N + 1 + 255) / 256), dim3(256), 0, stream>>>(
        row, row_ptr);

    gat_aggregate_kernel<<<dim3((GAT_N + 7) / 8), dim3(256), 0, stream>>>(
        hw, s_src, s_dst, row_ptr, col, out);
}
